// LayerNormLSTMCell_55903294325169
// MI455X (gfx1250) — matrix operand order repaired and checked by reading, not compiled, not run on hardware
//
#include <hip/hip_runtime.h>

// ---------------------------------------------------------------------------
// LayerNorm LSTM cell for MI455X (gfx1250), compile-only tuning loop.
//   B=4096, I=1024, H=1024  ->  fused GEMM  [B,2048] x [2048,4096] -> [B,4096]
//   bf16 WMMA (v_wmma_f32_16x16x32_bf16), fp32 accumulate.
//   Global->LDS staging via GLOBAL_LOAD_ASYNC_TO_LDS_B128 (ASYNCcnt),
//   double-buffered LDS, no VGPR staging (fixes round-1 scratch spills).
// ---------------------------------------------------------------------------

typedef __attribute__((ext_vector_type(16))) __bf16 v16bf;
typedef __attribute__((ext_vector_type(8)))  __bf16 bf16x8;
typedef __attribute__((ext_vector_type(8)))  float  v8f;
typedef __attribute__((ext_vector_type(8)))  unsigned short us8;

#define B_DIM 4096
#define I_DIM 1024
#define H_DIM 1024
#define K_DIM 2048   // I + H (fused K)
#define N_DIM 4096   // 4*H (gates)

#define BM 128
#define BN 128
#define BK 32
#define LDSP 40      // 32 shorts + 8 pad -> 80B row stride, conflict-free b128

__device__ __forceinline__ unsigned short f2bf(float f) {
  // round-to-nearest-even fp32 -> bf16
  unsigned int u = __float_as_uint(f);
  u += 0x7FFFu + ((u >> 16) & 1u);
  return (unsigned short)(u >> 16);
}

__device__ __forceinline__ float sigmoidf_(float x) {
  return 1.0f / (1.0f + __expf(-x));
}

// Async DMA: global (per-lane 64-bit addr) -> LDS (per-lane 32-bit offset).
// Generic LDS pointer's low 32 bits are the LDS byte offset (ISA 10.2).
__device__ __forceinline__ void async_copy_b128(void* lds_generic,
                                                const void* gptr) {
  unsigned int lds_off = (unsigned int)(unsigned long long)lds_generic;
  asm volatile("global_load_async_to_lds_b128 %0, %1, off"
               :
               : "v"(lds_off), "v"(gptr)
               : "memory");
}

__device__ __forceinline__ void wait_asynccnt0() {
  asm volatile("s_wait_asynccnt 0" ::: "memory");
}

// ---------------------------------------------------------------------------
// Pack two fp32 [4096,1024] matrices into one bf16 [4096,2048] row-major
// matrix (concatenated along K). Used for both (x|h) and (W_ih|W_hh).
// ---------------------------------------------------------------------------
__global__ __launch_bounds__(256) void pack_pair_bf16(
    const float* __restrict__ a, const float* __restrict__ b,
    unsigned short* __restrict__ dst) {
  int t    = blockIdx.x * 256 + threadIdx.x;
  int base = t * 8;                 // 8 elems per thread, never straddles 1024
  int row  = base >> 11;            // /2048
  int col  = base & 2047;
  const float* src = (col < I_DIM) ? (a + (size_t)row * I_DIM + col)
                                   : (b + (size_t)row * H_DIM + (col - I_DIM));
  float4 v0 = ((const float4*)src)[0];
  float4 v1 = ((const float4*)src)[1];
  us8 o;
  o[0] = f2bf(v0.x); o[1] = f2bf(v0.y); o[2] = f2bf(v0.z); o[3] = f2bf(v0.w);
  o[4] = f2bf(v1.x); o[5] = f2bf(v1.y); o[6] = f2bf(v1.z); o[7] = f2bf(v1.w);
  *(us8*)(dst + base) = o;
}

// ---------------------------------------------------------------------------
// gates[M,N] = A[M,K] * W[N,K]^T   (A = xh bf16, W = packed weights bf16)
// 256 threads = 8 waves; block tile 128x128; wave tile 32x64 (2x4 WMMA tiles)
// Double-buffered LDS fed by async DMA; one barrier per k-tile.
// ---------------------------------------------------------------------------
__global__ __launch_bounds__(256) void gemm_bf16_wmma(
    const unsigned short* __restrict__ A,
    const unsigned short* __restrict__ Bw,
    float* __restrict__ C) {
  __shared__ unsigned short As[2][BM][LDSP];
  __shared__ unsigned short Bs[2][BN][LDSP];

  const int tid   = threadIdx.x;
  const int lane  = tid & 31;
  const int wave  = tid >> 5;
  const int waveM = wave & 3;   // 4 waves along M -> 32-row strips
  const int waveN = wave >> 2;  // 2 waves along N -> 64-col strips
  const int rowA0 = blockIdx.y * BM;
  const int rowB0 = blockIdx.x * BN;

  // async-copy chunk mapping: tile = 128 rows x 64B; 512 b128 chunks/matrix;
  // thread t owns chunks t and t+256: row = c>>2, kcol = (c&3)*8 shorts.
  const int c0 = tid,        r0 = c0 >> 2, kc0 = (c0 & 3) * 8;
  const int c1 = tid + 256,  r1 = c1 >> 2, kc1 = (c1 & 3) * 8;

  const unsigned short* gA0 = A  + (size_t)(rowA0 + r0) * K_DIM + kc0;
  const unsigned short* gA1 = A  + (size_t)(rowA0 + r1) * K_DIM + kc1;
  const unsigned short* gB0 = Bw + (size_t)(rowB0 + r0) * K_DIM + kc0;
  const unsigned short* gB1 = Bw + (size_t)(rowB0 + r1) * K_DIM + kc1;

  v8f acc[2][4] = {};

  const int NK = K_DIM / BK;    // 64 k-tiles

  auto issue_tile = [&](int kt, int buf) {
    const int k0 = kt * BK;
    async_copy_b128(&As[buf][r0][kc0], gA0 + k0);
    async_copy_b128(&As[buf][r1][kc1], gA1 + k0);
    async_copy_b128(&Bs[buf][r0][kc0], gB0 + k0);
    async_copy_b128(&Bs[buf][r1][kc1], gB1 + k0);
    __builtin_prefetch(gA0 + k0 + 4 * BK, 0, 1);  // global_prefetch_b8, kt+4
    __builtin_prefetch(gB0 + k0 + 4 * BK, 0, 1);
  };

  // prologue: DMA k-tile 0 into buffer 0
  issue_tile(0, 0);
  wait_asynccnt0();
  __syncthreads();

  const int half = lane >> 4;     // 0: lanes 0-15, 1: lanes 16-31
  const int r16  = lane & 15;

  for (int kt = 0; kt < NK; ++kt) {
    const int cur = kt & 1;

    // kick off DMA for next k-tile into the other buffer
    if (kt + 1 < NK) issue_tile(kt + 1, cur ^ 1);

    // B fragments: lane = column N (r16), 32 contiguous K bytes
    // (lanes 0-15 -> K 0..15, lanes 16-31 -> K 16..31)
    v16bf bfr[4];
    #pragma unroll
    for (int wn = 0; wn < 4; ++wn) {
      const unsigned short* p = &Bs[cur][waveN * 64 + wn * 16 + r16][half * 16];
      bf16x8 lo = *(const bf16x8*)p;
      bf16x8 hi = *(const bf16x8*)(p + 8);
      #pragma unroll
      for (int e = 0; e < 8; ++e) { bfr[wn][e] = lo[e]; bfr[wn][e + 8] = hi[e]; }
    }
    // A fragments one at a time (keeps VGPR pressure low):
    // lane holds K 0..15 (+16 for upper half-wave), row M = r16
    #pragma unroll
    for (int wm = 0; wm < 2; ++wm) {
      const unsigned short* p = &As[cur][waveM * 32 + wm * 16 + r16][half * 16];
      bf16x8 lo = *(const bf16x8*)p;
      bf16x8 hi = *(const bf16x8*)(p + 8);
      v16bf af;
      #pragma unroll
      for (int e = 0; e < 8; ++e) { af[e] = lo[e]; af[e + 8] = hi[e]; }
      #pragma unroll
      for (int wn = 0; wn < 4; ++wn)
        acc[wm][wn] = __builtin_amdgcn_wmma_f32_16x16x32_bf16(
            false, af, false, bfr[wn], (short)0, acc[wm][wn], false, false);
    }

    // next buffer's DMA must have landed before anyone reads it
    if (kt + 1 < NK) wait_asynccnt0();
    __syncthreads();
  }

  // epilogue: C/D layout -> VGPR r holds M=r (+8 upper half-wave), N=lane&15
  const int rhalf = half * 8;
  #pragma unroll
  for (int wm = 0; wm < 2; ++wm) {
    #pragma unroll
    for (int wn = 0; wn < 4; ++wn) {
      int gcol = rowB0 + waveN * 64 + wn * 16 + r16;
      #pragma unroll
      for (int r = 0; r < 8; ++r) {
        int grow = rowA0 + waveM * 32 + wm * 16 + rhalf + r;
        C[(size_t)grow * N_DIM + gcol] = acc[wm][wn][r];
      }
    }
  }
}

// ---------------------------------------------------------------------------
// Per-row LayerNorm over 4H gates (+biases, pre-LN per reference), then LSTM
// pointwise update. One 256-thread block per batch row.
// ---------------------------------------------------------------------------
__global__ __launch_bounds__(256) void ln_lstm(
    const float* __restrict__ gates, const float* __restrict__ cprev,
    const float* __restrict__ b_ih, const float* __restrict__ b_hh,
    const float* __restrict__ gamma, const float* __restrict__ beta,
    float* __restrict__ out_h, float* __restrict__ out_c) {
  const int row = blockIdx.x;
  const int tid = threadIdx.x;
  const float* gr = gates + (size_t)row * N_DIM;

  float s = 0.f, ss = 0.f;
  #pragma unroll
  for (int j = 0; j < 16; ++j) {
    int idx = tid + 256 * j;
    float v = gr[idx] + b_ih[idx] + b_hh[idx];
    s += v; ss += v * v;
  }
  // wave32 butterfly reduce, then cross-wave via LDS
  #pragma unroll
  for (int off = 16; off > 0; off >>= 1) {
    s  += __shfl_xor(s,  off, 32);
    ss += __shfl_xor(ss, off, 32);
  }
  __shared__ float red[2][8];
  if ((tid & 31) == 0) { red[0][tid >> 5] = s; red[1][tid >> 5] = ss; }
  __syncthreads();
  float ts = 0.f, tss = 0.f;
  #pragma unroll
  for (int w = 0; w < 8; ++w) { ts += red[0][w]; tss += red[1][w]; }

  const float inv_n = 1.0f / (float)N_DIM;
  float mu  = ts * inv_n;
  float var = tss * inv_n - mu * mu;
  float rs  = __frsqrt_rn(var + 1e-5f);

  #pragma unroll
  for (int j = 0; j < 4; ++j) {
    int col = tid + 256 * j;   // H = 1024 = 4 * 256
    int i0 = col, i1 = col + H_DIM, i2 = col + 2 * H_DIM, i3 = col + 3 * H_DIM;
    float ig = (gr[i0] + b_ih[i0] + b_hh[i0] - mu) * rs * gamma[i0] + beta[i0];
    float fg = (gr[i1] + b_ih[i1] + b_hh[i1] - mu) * rs * gamma[i1] + beta[i1];
    float gg = (gr[i2] + b_ih[i2] + b_hh[i2] - mu) * rs * gamma[i2] + beta[i2];
    float og = (gr[i3] + b_ih[i3] + b_hh[i3] - mu) * rs * gamma[i3] + beta[i3];
    float cn = sigmoidf_(fg) * cprev[(size_t)row * H_DIM + col]
             + sigmoidf_(ig) * tanhf(gg);
    float hn = sigmoidf_(og) * tanhf(cn);
    out_h[(size_t)row * H_DIM + col] = hn;
    out_c[(size_t)row * H_DIM + col] = cn;
  }
}

// ---------------------------------------------------------------------------
extern "C" void kernel_launch(void* const* d_in, const int* in_sizes, int n_in,
                              void* d_out, int out_size, void* d_ws, size_t ws_size,
                              hipStream_t stream) {
  (void)in_sizes; (void)n_in; (void)out_size; (void)ws_size;

  const float* x     = (const float*)d_in[0];
  const float* h     = (const float*)d_in[1];
  const float* c     = (const float*)d_in[2];
  const float* w_ih  = (const float*)d_in[3];
  const float* w_hh  = (const float*)d_in[4];
  const float* b_ih  = (const float*)d_in[5];
  const float* b_hh  = (const float*)d_in[6];
  const float* gamma = (const float*)d_in[7];
  const float* beta  = (const float*)d_in[8];

  float* out_h = (float*)d_out;
  float* out_c = out_h + (size_t)B_DIM * H_DIM;

  // workspace layout: xh_bf16 (16MB) | w_bf16 (16MB) | gates_f32 (64MB)
  unsigned short* xh_bf = (unsigned short*)d_ws;
  unsigned short* w_bf  = xh_bf + (size_t)B_DIM * K_DIM;
  float*          gates = (float*)(w_bf + (size_t)N_DIM * K_DIM);

  // 1) pack activations and weights to bf16 (concat along K)
  int packBlocks = (B_DIM * K_DIM) / (256 * 8);   // 4096
  pack_pair_bf16<<<packBlocks, 256, 0, stream>>>(x, h, xh_bf);
  pack_pair_bf16<<<packBlocks, 256, 0, stream>>>(w_ih, w_hh, w_bf);

  // 2) fused gate GEMM: [4096,2048] x [2048,4096] (bf16 WMMA, f32 acc)
  dim3 grid(N_DIM / BN, B_DIM / BM);              // 32 x 32
  gemm_bf16_wmma<<<grid, 256, 0, stream>>>(xh_bf, w_bf, gates);

  // 3) LayerNorm + LSTM pointwise
  ln_lstm<<<B_DIM, 256, 0, stream>>>(gates, c, b_ih, b_hh, gamma, beta,
                                     out_h, out_c);
}
